// RiskPre_4544075399811
// MI455X (gfx1250) — matrix operand order repaired and checked by reading, not compiled, not run on hardware
//
#include <hip/hip_runtime.h>
#include <hip/hip_bf16.h>
#include <math.h>

typedef __attribute__((ext_vector_type(16))) __bf16 v16bf;
typedef __attribute__((ext_vector_type(8)))  float  v8f;

#define B_      512
#define T_      256
#define IN_DIM_ 128
#define H_DIM_  128
#define S_DIM_  64
#define GRU_    256
#define HID_    256
#define OUT_    256
#define NG_     5
#define NC_     2
#define BN_EPS_ 1e-5f

// ---------------- WMMA tile loaders (ISA 7.12.2 layouts, wave32) ----------------
// A: 16x32 bf16, row m = lane&15, hi = lane>>4; VGPR v: k = (v>>2)*16 + hi*8 + (v&3)*2
__device__ __forceinline__ v16bf load_a_tile(const __bf16* __restrict__ A, int lda,
                                             int m0, int kb, int lane) {
  int m  = m0 + (lane & 15);
  int hi = lane >> 4;
  const __bf16* p = A + (size_t)m * lda + kb;
  v16bf a;
#pragma unroll
  for (int v = 0; v < 8; ++v) {
    int k = ((v >> 2) << 4) + (hi << 3) + ((v & 3) << 1);
    a[2 * v]     = p[k];
    a[2 * v + 1] = p[k + 1];
  }
  return a;
}
// B: 32x16 bf16 where B(k,n) = W[n][k] (W row-major [N,K]); n = lane&15;
// VGPR v: k = (v>>2)*16 + hi*8 + (v&3)*2
__device__ __forceinline__ v16bf load_b_tile(const __bf16* __restrict__ W, int ldw,
                                             int n0, int kb, int lane) {
  int n  = n0 + (lane & 15);
  int hi = lane >> 4;
  const __bf16* p = W + (size_t)n * ldw + kb;
  v16bf b;
#pragma unroll
  for (int v = 0; v < 8; ++v) {
    int k = ((v >> 2) << 4) + (hi << 3) + ((v & 3) << 1);
    b[2 * v]     = p[k];
    b[2 * v + 1] = p[k + 1];
  }
  return b;
}

#define WMMA_BF16(a, b, c) \
  __builtin_amdgcn_wmma_f32_16x16x32_bf16(false, (a), false, (b), (short)0, (c), false, false)

// ---------------- deterministic RNG (stand-in for jax threefry) ----------------
__device__ __forceinline__ unsigned hash_u32(unsigned x) {
  x ^= x >> 17; x *= 0xed5ad4bbu;
  x ^= x >> 11; x *= 0xac4c1b51u;
  x ^= x >> 15; x *= 0x31848babu;
  x ^= x >> 14;
  return x;
}
__device__ __forceinline__ float uniform01(unsigned seed, unsigned idx) {
  unsigned h = hash_u32(seed * 0x9e3779b9u + idx * 2u + 1u);
  return (float)(h >> 8) * (1.0f / 16777216.0f) + 1e-7f;
}
__device__ __forceinline__ float rand_normal(unsigned seed, unsigned idx) {
  float u1 = uniform01(seed, idx * 2u);
  float u2 = uniform01(seed ^ 0x85ebca6bu, idx * 2u + 1u);
  return sqrtf(-2.0f * logf(u1)) * cosf(6.28318530718f * u2);
}
__device__ __forceinline__ float sigmoidf_(float x) { return 1.0f / (1.0f + expf(-x)); }
__device__ __forceinline__ float elup1_(float x) {
  return (x > 0.0f ? x : expm1f(x)) + 1.0f + 1e-12f;
}

// ---------------- conversion / init kernels ----------------
__global__ void k_cvt_bf16(__bf16* dst, const float* src, int n) {
  for (int i = blockIdx.x * 256 + threadIdx.x; i < n; i += gridDim.x * 256)
    dst[i] = (__bf16)src[i];
}
__global__ void k_bcast_h0(const float* h0, float* h_f, __bf16* h_bf) {
  int idx = blockIdx.x * 256 + threadIdx.x;  // 512*256 threads
  int j = idx & (GRU_ - 1);
  float v = h0[j];
  h_f[idx]  = v;
  h_bf[idx] = (__bf16)v;
}
__global__ void k_init_hq(const float* hq0, __bf16* hprev_bf, __bf16* h_all_bf) {
  int idx = blockIdx.x * 256 + threadIdx.x;  // 512*128 threads
  int b = idx >> 7, j = idx & 127;
  __bf16 v = (__bf16)hq0[j];
  hprev_bf[idx] = v;
  h_all_bf[(size_t)b * T_ * H_DIM_ + j] = v;  // t = 0 column
}

// ---------------- generic WMMA GEMM: C[M,N] = act(A[M,K] @ W[N,K]^T + bias) ----------------
__global__ void k_gemm_bf16(const __bf16* __restrict__ A, int lda,
                            const __bf16* __restrict__ W, int ldw,
                            const float* __restrict__ bias,
                            float* __restrict__ C, int ldc,
                            int Ntiles, int K, int relu) {
  int lane = threadIdx.x & 31;
  int tile = blockIdx.x * 4 + (threadIdx.x >> 5);
  int m0 = (tile / Ntiles) * 16;
  int n0 = (tile % Ntiles) * 16;
  const __bf16* pfA = A + (size_t)(m0 + (lane & 15)) * lda;
  const __bf16* pfW = W + (size_t)(n0 + (lane & 15)) * ldw;
  v8f acc = {};
  for (int kb = 0; kb < K; kb += 32) {
    // prefetch next K-chunk one iteration ahead (global_prefetch_b8); uniform branch
    if (kb + 32 < K) {
      __builtin_prefetch(pfA + kb + 32, 0, 1);
      __builtin_prefetch(pfW + kb + 32, 0, 1);
    }
    v16bf a = load_a_tile(A, lda, m0, kb, lane);
    v16bf b = load_b_tile(W, ldw, n0, kb, lane);
    acc = WMMA_BF16(a, b, acc);
  }
  int hi = lane >> 4;
  int n  = n0 + (lane & 15);
  float bv = bias[n];
#pragma unroll
  for (int r = 0; r < 8; ++r) {
    int m = m0 + hi * 8 + r;
    float v = acc[r] + bv;
    if (relu) v = fmaxf(v, 0.0f);
    C[(size_t)m * ldc + n] = v;
  }
}

// ---------------- fused GRU step: gates from [x_t | h] @ [W_ih ; W_hh]^T ----------------
__global__ void k_gru_step(int t, const __bf16* __restrict__ x_bf,
                           const __bf16* __restrict__ h_in_bf,
                           const float* __restrict__ h_in_f,
                           const __bf16* __restrict__ W_ih_bf,
                           const __bf16* __restrict__ W_hh_bf,
                           const float* __restrict__ b_ih,
                           const float* __restrict__ b_hh,
                           __bf16* __restrict__ h_out_bf,
                           float* __restrict__ h_out_f,
                           __bf16* __restrict__ h_gru_bf) {
  int lane = threadIdx.x & 31;
  int tile = blockIdx.x * 4 + (threadIdx.x >> 5);  // 512 tiles = 32 m-tiles x 16 n-tiles
  int m0 = (tile >> 4) * 16;
  int n0 = (tile & 15) * 16;
  v8f ar = {}, az = {}, axn = {}, ahn = {};
  for (int kb = 0; kb < IN_DIM_ + GRU_; kb += 32) {  // K = 384, split at 128 (mult of 32)
    v16bf a;
    if (kb < IN_DIM_)
      a = load_a_tile(x_bf + (size_t)t * IN_DIM_, T_ * IN_DIM_, m0, kb, lane);
    else
      a = load_a_tile(h_in_bf, GRU_, m0, kb - IN_DIM_, lane);
    if (kb < IN_DIM_) {
      v16bf br = load_b_tile(W_ih_bf,                           IN_DIM_, n0, kb, lane);
      v16bf bz = load_b_tile(W_ih_bf + (size_t)GRU_ * IN_DIM_,  IN_DIM_, n0, kb, lane);
      v16bf bn = load_b_tile(W_ih_bf + (size_t)2*GRU_*IN_DIM_,  IN_DIM_, n0, kb, lane);
      ar  = WMMA_BF16(a, br, ar);
      az  = WMMA_BF16(a, bz, az);
      axn = WMMA_BF16(a, bn, axn);   // x-part of n-gate kept separate
    } else {
      int kh = kb - IN_DIM_;
      // prefetch next h-chunk of the recurrent weights (uniform branch)
      if (kb + 32 < IN_DIM_ + GRU_) {
        __builtin_prefetch(W_hh_bf + (size_t)(n0 + (lane & 15)) * GRU_ + kh + 32, 0, 1);
        __builtin_prefetch(h_in_bf + (size_t)(m0 + (lane & 15)) * GRU_ + kh + 32, 0, 1);
      }
      v16bf br = load_b_tile(W_hh_bf,                          GRU_, n0, kh, lane);
      v16bf bz = load_b_tile(W_hh_bf + (size_t)GRU_ * GRU_,    GRU_, n0, kh, lane);
      v16bf bn = load_b_tile(W_hh_bf + (size_t)2*GRU_*GRU_,    GRU_, n0, kh, lane);
      ar  = WMMA_BF16(a, br, ar);
      az  = WMMA_BF16(a, bz, az);
      ahn = WMMA_BF16(a, bn, ahn);   // h-part of n-gate (gets multiplied by r)
    }
  }
  int hi = lane >> 4;
  int n  = n0 + (lane & 15);
  float br_b = b_ih[n]            + b_hh[n];
  float bz_b = b_ih[GRU_ + n]     + b_hh[GRU_ + n];
  float bxn  = b_ih[2 * GRU_ + n];
  float bhn  = b_hh[2 * GRU_ + n];
#pragma unroll
  for (int r = 0; r < 8; ++r) {
    int m = m0 + hi * 8 + r;
    float rg = sigmoidf_(ar[r] + br_b);
    float zg = sigmoidf_(az[r] + bz_b);
    float ng = tanhf(axn[r] + bxn + rg * (ahn[r] + bhn));
    float ho = h_in_f[(size_t)m * GRU_ + n];
    float hn = (1.0f - zg) * ng + zg * ho;
    h_out_f[(size_t)m * GRU_ + n]  = hn;
    h_out_bf[(size_t)m * GRU_ + n] = (__bf16)hn;
    h_gru_bf[((size_t)t * B_ + m) * GRU_ + n] = (__bf16)hn;
  }
}

// ---------------- batchnorm stats: biased mean/var over batch, per feature ----------------
__global__ void k_bn_stats(const float* __restrict__ src, float* mean, float* rstd) {
  __shared__ float ssum[128], ssq[128];
  int f = blockIdx.x, tid = threadIdx.x;
  float s = 0.0f, q = 0.0f;
  for (int b = tid; b < B_; b += 128) {
    float v = src[(size_t)b * HID_ + f];
    s += v; q += v * v;
  }
  ssum[tid] = s; ssq[tid] = q;
  __syncthreads();
  for (int off = 64; off > 0; off >>= 1) {
    if (tid < off) { ssum[tid] += ssum[tid + off]; ssq[tid] += ssq[tid + off]; }
    __syncthreads();
  }
  if (tid == 0) {
    float m = ssum[0] * (1.0f / B_);
    float var = ssq[0] * (1.0f / B_) - m * m;
    mean[f] = m;
    rstd[f] = rsqrtf(var + BN_EPS_);
  }
}

// ---------------- he = [BN1(pre1) ; s] in bf16 ----------------
__global__ void k_make_he(const float* __restrict__ pre1, const float* mean1,
                          const float* rstd1, const float* g1, const float* be1,
                          const float* __restrict__ s, __bf16* __restrict__ he_bf) {
  int idx = blockIdx.x * 256 + threadIdx.x;  // 512*320
  int b = idx / (HID_ + S_DIM_);
  int j = idx - b * (HID_ + S_DIM_);
  float v;
  if (j < HID_)
    v = g1[j] * (pre1[(size_t)b * HID_ + j] - mean1[j]) * rstd1[j] + be1[j];
  else
    v = s[(size_t)b * S_DIM_ + (j - HID_)];
  he_bf[idx] = (__bf16)v;
}

// ---------------- hc = 0.5*tanh(BN2(pre2) + hg_t) in bf16 ----------------
__global__ void k_make_hc(int t, const float* __restrict__ pre2, const float* mean2,
                          const float* rstd2, const float* g2, const float* be2,
                          const __bf16* __restrict__ h_gru_bf, __bf16* __restrict__ hc_bf) {
  int idx = blockIdx.x * 256 + threadIdx.x;  // 512*256
  int b = idx >> 8, k = idx & 255;
  float v = g2[k] * (pre2[idx] - mean2[k]) * rstd2[k] + be2[k]
          + (float)h_gru_bf[((size_t)t * B_ + b) * GRU_ + k];
  hc_bf[idx] = (__bf16)(0.5f * tanhf(v));
}

// ---------------- ht = mu + (elu(sp)+1)*eps ; write h_all (bf16) and hprev ----------------
__global__ void k_reparam(int t, const float* __restrict__ musd,
                          __bf16* __restrict__ h_all_bf, __bf16* __restrict__ hprev_bf) {
  int idx = blockIdx.x * 256 + threadIdx.x;  // 512*128
  int b = idx >> 7, j = idx & 127;
  float mu  = musd[(size_t)b * 256 + j];
  float sp  = musd[(size_t)b * 256 + 128 + j];
  float std = elup1_(sp);
  float e   = rand_normal(42u, (unsigned)(((t - 1) * B_ + b) * H_DIM_ + j));
  float ht  = mu + std * e;
  h_all_bf[(size_t)b * T_ * H_DIM_ + (size_t)t * H_DIM_ + j] = (__bf16)ht;
  hprev_bf[idx] = (__bf16)ht;
}

// ---------------- MDN head: gumbel-argmax pick, reparam, softmax over NC=2 ----------------
__global__ void k_mdn_final(const float* __restrict__ he2, const float* __restrict__ s,
                            const float* __restrict__ Wpi, const float* __restrict__ bpi,
                            const float* __restrict__ Wmu, const float* __restrict__ bmu,
                            const float* __restrict__ Wsg, const float* __restrict__ bsg,
                            float* __restrict__ out) {
  int b = blockIdx.x * 256 + threadIdx.x;
  if (b >= B_) return;
  const float* hb = he2 + (size_t)b * OUT_;
  const float* sb = s + (size_t)b * S_DIM_;
  float best = -1e30f; int pi = 0;
  for (int g = 0; g < NG_; ++g) {
    const float* w = Wpi + (size_t)g * (OUT_ + S_DIM_);
    float acc = bpi[g];
    for (int k = 0; k < OUT_; ++k)   acc += hb[k] * w[k];
    for (int k = 0; k < S_DIM_; ++k) acc += sb[k] * w[OUT_ + k];
    float u = uniform01(7u, (unsigned)(b * NG_ + g));
    float v = acc - logf(-logf(u));  // gumbel-argmax categorical
    if (v > best) { best = v; pi = g; }
  }
  float z[NC_];
  for (int c = 0; c < NC_; ++c) {
    int row = c * NG_ + pi;
    const float* wm = Wmu + (size_t)row * (OUT_ + S_DIM_);
    const float* wg = Wsg + (size_t)row * (OUT_ + S_DIM_);
    float am = bmu[row], ag = bsg[row];
    for (int k = 0; k < OUT_; ++k)   { float h = hb[k]; am += h * wm[k]; ag += h * wg[k]; }
    for (int k = 0; k < S_DIM_; ++k) { float v = sb[k]; am += v * wm[OUT_ + k]; ag += v * wg[OUT_ + k]; }
    float sg = elup1_(ag);
    float e  = rand_normal(9u, (unsigned)(b * NC_ + c));
    z[c] = am + sg * e;
  }
  float mx = fmaxf(z[0], z[1]);
  float e0 = expf(z[0] - mx), e1 = expf(z[1] - mx);
  float inv = 1.0f / (e0 + e1);
  out[b * 2 + 0] = e0 * inv;
  out[b * 2 + 1] = e1 * inv;
}

// ======================================================================================
extern "C" void kernel_launch(void* const* d_in, const int* in_sizes, int n_in,
                              void* d_out, int out_size, void* d_ws, size_t ws_size,
                              hipStream_t stream) {
  const float* x    = (const float*)d_in[0];
  const float* s    = (const float*)d_in[1];
  const float* W_ih = (const float*)d_in[2];
  const float* W_hh = (const float*)d_in[3];
  const float* b_ih = (const float*)d_in[4];
  const float* b_hh = (const float*)d_in[5];
  const float* h0   = (const float*)d_in[6];
  const float* hq0  = (const float*)d_in[7];
  const float* W1   = (const float*)d_in[8];
  const float* b1   = (const float*)d_in[9];
  const float* g1   = (const float*)d_in[10];
  const float* be1  = (const float*)d_in[11];
  const float* W2   = (const float*)d_in[12];
  const float* b2   = (const float*)d_in[13];
  const float* g2   = (const float*)d_in[14];
  const float* be2  = (const float*)d_in[15];
  const float* Wm   = (const float*)d_in[16];
  const float* bm   = (const float*)d_in[17];
  const float* Wsd  = (const float*)d_in[18];
  const float* bsd  = (const float*)d_in[19];
  const float* Wp   = (const float*)d_in[20];
  const float* bp   = (const float*)d_in[21];
  const float* Wpi  = (const float*)d_in[22];
  const float* bpi  = (const float*)d_in[23];
  const float* Wmu  = (const float*)d_in[24];
  const float* bmu  = (const float*)d_in[25];
  const float* Wsg  = (const float*)d_in[26];
  const float* bsg  = (const float*)d_in[27];
  float* out = (float*)d_out;

  // ---- carve workspace (~156 MB) ----
  char* p = (char*)d_ws;
  auto carve = [&](size_t bytes) -> char* {
    char* r = p;
    p += (bytes + 255) & ~(size_t)255;
    return r;
  };
  __bf16* x_bf     = (__bf16*)carve((size_t)B_ * T_ * IN_DIM_ * 2);     // 33.5 MB
  __bf16* W_ih_bf  = (__bf16*)carve((size_t)3 * GRU_ * IN_DIM_ * 2);
  __bf16* W_hh_bf  = (__bf16*)carve((size_t)3 * GRU_ * GRU_ * 2);
  __bf16* W1_bf    = (__bf16*)carve((size_t)HID_ * H_DIM_ * 2);
  __bf16* W2_bf    = (__bf16*)carve((size_t)GRU_ * (HID_ + S_DIM_) * 2);
  __bf16* Wms_bf   = (__bf16*)carve((size_t)2 * H_DIM_ * GRU_ * 2);     // [Wm;Ws] stacked
  __bf16* Wp_bf    = (__bf16*)carve((size_t)OUT_ * H_DIM_ * T_ * 2);    // 16.8 MB
  __bf16* h_gru_bf = (__bf16*)carve((size_t)T_ * B_ * GRU_ * 2);        // 67 MB
  __bf16* h_all_bf = (__bf16*)carve((size_t)B_ * T_ * H_DIM_ * 2);      // 33.5 MB
  float*  hA_f     = (float*)carve((size_t)B_ * GRU_ * 4);
  float*  hB_f     = (float*)carve((size_t)B_ * GRU_ * 4);
  __bf16* hA_bf    = (__bf16*)carve((size_t)B_ * GRU_ * 2);
  __bf16* hB_bf    = (__bf16*)carve((size_t)B_ * GRU_ * 2);
  __bf16* hprev_bf = (__bf16*)carve((size_t)B_ * H_DIM_ * 2);
  float*  pre1     = (float*)carve((size_t)B_ * HID_ * 4);
  float*  pre2     = (float*)carve((size_t)B_ * GRU_ * 4);
  __bf16* he_bf    = (__bf16*)carve((size_t)B_ * (HID_ + S_DIM_) * 2);
  __bf16* hc_bf    = (__bf16*)carve((size_t)B_ * GRU_ * 2);
  float*  musd     = (float*)carve((size_t)B_ * 256 * 4);
  float*  he2      = (float*)carve((size_t)B_ * OUT_ * 4);
  float*  bias_ms  = (float*)carve(256 * 4);
  float*  mean1    = (float*)carve(HID_ * 4);
  float*  rstd1    = (float*)carve(HID_ * 4);
  float*  mean2    = (float*)carve(GRU_ * 4);
  float*  rstd2    = (float*)carve(GRU_ * 4);

  // ---- stage 0: bf16 conversions (weights resident in L2 afterwards) ----
  k_cvt_bf16<<<2048, 256, 0, stream>>>(x_bf, x, B_ * T_ * IN_DIM_);
  k_cvt_bf16<<<96, 256, 0, stream>>>(W_ih_bf, W_ih, 3 * GRU_ * IN_DIM_);
  k_cvt_bf16<<<192, 256, 0, stream>>>(W_hh_bf, W_hh, 3 * GRU_ * GRU_);
  k_cvt_bf16<<<64, 256, 0, stream>>>(W1_bf, W1, HID_ * H_DIM_);
  k_cvt_bf16<<<160, 256, 0, stream>>>(W2_bf, W2, GRU_ * (HID_ + S_DIM_));
  k_cvt_bf16<<<64, 256, 0, stream>>>(Wms_bf, Wm, H_DIM_ * GRU_);
  k_cvt_bf16<<<64, 256, 0, stream>>>(Wms_bf + (size_t)H_DIM_ * GRU_, Wsd, H_DIM_ * GRU_);
  k_cvt_bf16<<<2048, 256, 0, stream>>>(Wp_bf, Wp, OUT_ * H_DIM_ * T_);
  hipMemcpyAsync(bias_ms, bm, H_DIM_ * sizeof(float), hipMemcpyDeviceToDevice, stream);
  hipMemcpyAsync(bias_ms + H_DIM_, bsd, H_DIM_ * sizeof(float), hipMemcpyDeviceToDevice, stream);

  k_bcast_h0<<<(B_ * GRU_) / 256, 256, 0, stream>>>(h0, hA_f, hA_bf);
  k_init_hq<<<(B_ * H_DIM_) / 256, 256, 0, stream>>>(hq0, hprev_bf, h_all_bf);

  // ---- stage 1: GRU scan, one fused WMMA kernel per step (weights hot in L2) ----
  for (int t = 0; t < T_; ++t) {
    const __bf16* hin_bf = (t & 1) ? hB_bf : hA_bf;
    const float*  hin_f  = (t & 1) ? hB_f  : hA_f;
    __bf16* hout_bf = (t & 1) ? hA_bf : hB_bf;
    float*  hout_f  = (t & 1) ? hA_f  : hB_f;
    k_gru_step<<<128, 128, 0, stream>>>(t, x_bf, hin_bf, hin_f, W_ih_bf, W_hh_bf,
                                        b_ih, b_hh, hout_bf, hout_f, h_gru_bf);
  }

  // ---- stage 2: sequential inference recurrence (BatchNorm forces per-step sync) ----
  for (int t = 1; t < T_; ++t) {
    // pre1 = relu(hprev @ W1^T + b1): M=512 N=256 K=128
    k_gemm_bf16<<<128, 128, 0, stream>>>(hprev_bf, H_DIM_, W1_bf, H_DIM_, b1,
                                         pre1, HID_, HID_ / 16, H_DIM_, 1);
    k_bn_stats<<<HID_, 128, 0, stream>>>(pre1, mean1, rstd1);
    k_make_he<<<(B_ * (HID_ + S_DIM_)) / 256, 256, 0, stream>>>(pre1, mean1, rstd1,
                                                                g1, be1, s, he_bf);
    // pre2 = relu(he @ W2^T + b2): M=512 N=256 K=320
    k_gemm_bf16<<<128, 128, 0, stream>>>(he_bf, HID_ + S_DIM_, W2_bf, HID_ + S_DIM_, b2,
                                         pre2, GRU_, GRU_ / 16, HID_ + S_DIM_, 1);
    k_bn_stats<<<GRU_, 128, 0, stream>>>(pre2, mean2, rstd2);
    k_make_hc<<<(B_ * GRU_) / 256, 256, 0, stream>>>(t, pre2, mean2, rstd2, g2, be2,
                                                     h_gru_bf, hc_bf);
    // [mu ; std_pre] = hc @ [Wm;Ws]^T + [bm;bs]: M=512 N=256 K=256
    k_gemm_bf16<<<128, 128, 0, stream>>>(hc_bf, GRU_, Wms_bf, GRU_, bias_ms,
                                         musd, 256, 256 / 16, GRU_, 0);
    k_reparam<<<(B_ * H_DIM_) / 256, 256, 0, stream>>>(t, musd, h_all_bf, hprev_bf);
  }

  // ---- stage 3: MDN head. he2 = relu(h_all @ Wp^T + bp): M=512 N=256 K=32768 ----
  k_gemm_bf16<<<128, 128, 0, stream>>>(h_all_bf, T_ * H_DIM_, Wp_bf, T_ * H_DIM_, bp,
                                       he2, OUT_, OUT_ / 16, T_ * H_DIM_, 1);
  k_mdn_final<<<2, 256, 0, stream>>>(he2, s, Wpi, bpi, Wmu, bmu, Wsg, bsg, out);
}
